// TransitionModel_80659485819129
// MI455X (gfx1250) — matrix operand order repaired and checked by reading, compiled, not run on hardware
//
#include <hip/hip_runtime.h>
#include <hip/hip_bf16.h>
#include <math.h>

// ---------------- problem constants (from reference) ----------------
#define TT 50
#define BB 512
#define SS 256
#define AA 32
#define BEL 1024
#define HH 1024
#define EE 1024
#define MM 256
#define KSA (SS + AA + MM)     // 544
#define KBQ (BEL + EE + MM)    // 2304
#define MIN_STD 0.1f

// ---------------- types ----------------
typedef __attribute__((ext_vector_type(16))) __bf16 v16bf;
typedef __attribute__((ext_vector_type(8)))  __bf16 v8bf;
typedef __attribute__((ext_vector_type(8)))  float  v8f;

__device__ __forceinline__ __bf16 f2bf(float f) {
  union { float f; unsigned u; } v; v.f = f;
  unsigned u = v.u;
  unsigned r = 0x7FFFu + ((u >> 16) & 1u);
  unsigned short h = (unsigned short)((u + r) >> 16);
  __bf16 out;
  __builtin_memcpy(&out, &h, 2);
  return out;
}

__device__ __forceinline__ float sigmoidf_(float x) { return 1.0f / (1.0f + __expf(-x)); }
__device__ __forceinline__ float softplusf_(float x) {
  // stable: max(x,0) + log1p(exp(-|x|))
  return fmaxf(x, 0.0f) + log1pf(__expf(-fabsf(x)));
}

// ---------------- WMMA fragment loaders (ISA 7.12.2 layouts) ----------------
// A (16x32 bf16): lane holds row m = lane%16; element j -> K = (j%8) + 16*(j/8) + 8*(lane/16)
__device__ __forceinline__ v16bf ldA(const __bf16* p /* row base + k0 + 8*(lane>>4) */) {
  v8bf lo = *(const v8bf*)(p);
  v8bf hi = *(const v8bf*)(p + 16);
  v16bf r;
#pragma unroll
  for (int j = 0; j < 8; ++j) { r[j] = lo[j]; r[8 + j] = hi[j]; }
  return r;
}
// B (32x16 bf16): lane holds column n = lane%16; element j -> K = (j%8) + 16*(j/8) + 8*(lane/16)
__device__ __forceinline__ v16bf ldB(const __bf16* p /* W row n + k0 + 8*(lane>>4) */) {
  v8bf lo = *(const v8bf*)(p);
  v8bf hi = *(const v8bf*)(p + 16);
  v16bf r;
#pragma unroll
  for (int j = 0; j < 8; ++j) { r[j] = lo[j]; r[8 + j] = hi[j]; }
  return r;
}

// ---------------- bf16 WMMA GEMM: out[M,N] = X[M,K] @ W[N,K]^T + bias ----------------
// Block: 128 threads = 4 waves; wave tile 32x32; block tile 32x128.
// grid = (N/128, M/32). Exact tiling only (no bounds checks -> EXEC all ones).
template <int RELU, int OUTF, int OUTB>
__global__ __launch_bounds__(128) void gemm_wmma_k(
    const __bf16* __restrict__ Xb, const __bf16* __restrict__ Wb,
    const float* __restrict__ bias, float* __restrict__ outF,
    __bf16* __restrict__ outB, int K, int N) {
  const int lane = threadIdx.x & 31;
  const int wave = threadIdx.x >> 5;
  const int row0 = blockIdx.y * 32;
  const int col0 = blockIdx.x * 128 + wave * 32;
  const int mlo = lane & 15;
  const int hif = lane >> 4;

  const __bf16* xr0 = Xb + (size_t)(row0 + mlo) * K + hif * 8;
  const __bf16* xr1 = xr0 + (size_t)16 * K;
  const __bf16* wr0 = Wb + (size_t)(col0 + mlo) * K + hif * 8;
  const __bf16* wr1 = wr0 + (size_t)16 * K;

  v8f acc00 = {}, acc01 = {}, acc10 = {}, acc11 = {};

  for (int k0 = 0; k0 < K; k0 += 32) {
    v16bf a0 = ldA(xr0 + k0);
    v16bf a1 = ldA(xr1 + k0);
    v16bf b0 = ldB(wr0 + k0);
    v16bf b1 = ldB(wr1 + k0);
    acc00 = __builtin_amdgcn_wmma_f32_16x16x32_bf16(false, a0, false, b0, (short)0, acc00, false, false);
    acc01 = __builtin_amdgcn_wmma_f32_16x16x32_bf16(false, a0, false, b1, (short)0, acc01, false, false);
    acc10 = __builtin_amdgcn_wmma_f32_16x16x32_bf16(false, a1, false, b0, (short)0, acc10, false, false);
    acc11 = __builtin_amdgcn_wmma_f32_16x16x32_bf16(false, a1, false, b1, (short)0, acc11, false, false);
  }

  const int n0 = col0 + mlo;
  const float bi0 = bias[n0];
  const float bi1 = bias[n0 + 16];
  const int rb = row0 + hif * 8;
#pragma unroll
  for (int r = 0; r < 8; ++r) {
    float v00 = acc00[r] + bi0, v01 = acc01[r] + bi1;
    float v10 = acc10[r] + bi0, v11 = acc11[r] + bi1;
    if (RELU) {
      v00 = fmaxf(v00, 0.f); v01 = fmaxf(v01, 0.f);
      v10 = fmaxf(v10, 0.f); v11 = fmaxf(v11, 0.f);
    }
    const size_t o0 = (size_t)(rb + r) * N;
    const size_t o1 = (size_t)(rb + 16 + r) * N;
    if (OUTF) {
      outF[o0 + n0] = v00; outF[o0 + n0 + 16] = v01;
      outF[o1 + n0] = v10; outF[o1 + n0 + 16] = v11;
    }
    if (OUTB) {
      outB[o0 + n0] = f2bf(v00); outB[o0 + n0 + 16] = f2bf(v01);
      outB[o1 + n0] = f2bf(v10); outB[o1 + n0 + 16] = f2bf(v11);
    }
  }
}

// ---------------- elementwise helper kernels ----------------
__global__ void cvt_bf16_k(const float* __restrict__ in, __bf16* __restrict__ out, int n) {
  int i = blockIdx.x * blockDim.x + threadIdx.x;
  if (i < n) out[i] = f2bf(in[i]);
}

__global__ void init_state_k(const float* __restrict__ ps, float* __restrict__ st, int n) {
  int i = blockIdx.x * blockDim.x + threadIdx.x;
  if (i < n) st[i] = ps[i];
}

__global__ void init_belief_k(const float* __restrict__ pb, float* __restrict__ bf,
                              __bf16* __restrict__ bb, int n) {
  int i = blockIdx.x * blockDim.x + threadIdx.x;
  if (i < n) { float v = pb[i]; bf[i] = v; bb[i] = f2bf(v); }
}

// X_sa[b, :] = bf16([state*nt, action, map_emb])
__global__ void pack_sa_k(const float* __restrict__ st, const float* __restrict__ nt,
                          const float* __restrict__ act, const float* __restrict__ me,
                          __bf16* __restrict__ Xsa) {
  int i = blockIdx.x * blockDim.x + threadIdx.x;
  if (i >= BB * KSA) return;
  int b = i / KSA, c = i % KSA;
  float v;
  if (c < SS)            v = st[b * SS + c] * nt[b];
  else if (c < SS + AA)  v = act[b * AA + (c - SS)];
  else                   v = me[b * MM + (c - SS - AA)];
  Xsa[i] = f2bf(v);
}

// X_bq[b, BEL: ] = bf16([obs, map_emb_post])  (belief part written by gru_gates_k)
__global__ void pack_bq_k(const float* __restrict__ obs, const float* __restrict__ mp,
                          __bf16* __restrict__ Xbq) {
  int i = blockIdx.x * blockDim.x + threadIdx.x;
  if (i >= BB * (EE + MM)) return;
  int b = i / (EE + MM), c = i % (EE + MM);
  if (c < EE) Xbq[(size_t)b * KBQ + BEL + c] = f2bf(obs[b * EE + c]);
  else        Xbq[(size_t)b * KBQ + BEL + EE + (c - EE)] = f2bf(mp[b * MM + (c - EE)]);
}

// GRUCell gates + belief update; also scatters belief into X_bq and d_out
__global__ void gru_gates_k(const float* __restrict__ gi, const float* __restrict__ gh,
                            float* __restrict__ belf, __bf16* __restrict__ belb,
                            __bf16* __restrict__ Xbq, float* __restrict__ out_belief) {
  int i = blockIdx.x * blockDim.x + threadIdx.x;
  if (i >= BB * BEL) return;
  int b = i / BEL, j = i % BEL;
  size_t base = (size_t)b * (3 * BEL) + j;
  float ir = gi[base], iz = gi[base + BEL], in_ = gi[base + 2 * BEL];
  float hr = gh[base], hz = gh[base + BEL], hn = gh[base + 2 * BEL];
  float r = sigmoidf_(ir + hr);
  float z = sigmoidf_(iz + hz);
  float n = tanhf(in_ + r * hn);
  float h = belf[i];
  float nb = (1.0f - z) * n + z * h;
  belf[i] = nb;
  __bf16 nbb = f2bf(nb);
  belb[i] = nbb;
  Xbq[(size_t)b * KBQ + j] = nbb;
  out_belief[i] = nb;
}

// split mean/raw, std = min(softplus(raw)+MIN_STD, 5), state = mean + std*noise
template <int WRITE_STATE>
__global__ void dist_out_k(const float* __restrict__ p2, const float* __restrict__ noise,
                           float* __restrict__ oMean, float* __restrict__ oStd,
                           float* __restrict__ oState, float* __restrict__ stbuf) {
  int i = blockIdx.x * blockDim.x + threadIdx.x;
  if (i >= BB * SS) return;
  int b = i / SS, j = i % SS;
  float mean = p2[(size_t)b * (2 * SS) + j];
  float raw  = p2[(size_t)b * (2 * SS) + SS + j];
  float sd = fminf(softplusf_(raw) + MIN_STD, 5.0f);
  float st = mean + sd * noise[i];
  oMean[i] = mean; oStd[i] = sd; oState[i] = st;
  if (WRITE_STATE) stbuf[i] = st;
}

// ---------------- host-side launch ----------------
extern "C" void kernel_launch(void* const* d_in, const int* in_sizes, int n_in,
                              void* d_out, int out_size, void* d_ws, size_t ws_size,
                              hipStream_t stream) {
  const float* prev_state = (const float*)d_in[0];
  const float* actions    = (const float*)d_in[1];
  const float* prev_bel   = (const float*)d_in[2];
  const float* obs        = (const float*)d_in[3];
  const float* nonterm    = (const float*)d_in[4];
  const float* map_e      = (const float*)d_in[5];
  const float* map_ep     = (const float*)d_in[6];
  const float* noi_p      = (const float*)d_in[7];
  const float* noi_q      = (const float*)d_in[8];
  const float* W_sa = (const float*)d_in[9];  const float* b_sa = (const float*)d_in[10];
  const float* W_ih = (const float*)d_in[11]; const float* b_ih = (const float*)d_in[12];
  const float* W_hh = (const float*)d_in[13]; const float* b_hh = (const float*)d_in[14];
  const float* W_bp = (const float*)d_in[15]; const float* b_bp = (const float*)d_in[16];
  const float* W_sp = (const float*)d_in[17]; const float* b_sp = (const float*)d_in[18];
  const float* W_bq = (const float*)d_in[19]; const float* b_bq = (const float*)d_in[20];
  const float* W_sq = (const float*)d_in[21]; const float* b_sq = (const float*)d_in[22];

  float* out = (float*)d_out;
  const size_t oBelief = 0;
  const size_t nB = (size_t)TT * BB * BEL;   // belief elements
  const size_t nS = (size_t)TT * BB * SS;    // per state-array elements
  float* out_belief = out + oBelief;
  float* out_pstate = out + nB;
  float* out_pmean  = out + nB + 1 * nS;
  float* out_pstd   = out + nB + 2 * nS;
  float* out_qstate = out + nB + 3 * nS;
  float* out_qmean  = out + nB + 4 * nS;
  float* out_qstd   = out + nB + 5 * nS;

  // ---- workspace carve-out (all sizes are multiples of 256 B) ----
  char* ws = (char*)d_ws;
  size_t cur = 0;
  auto carve = [&](size_t bytes) { char* p = ws + cur; cur += (bytes + 255) & ~(size_t)255; return p; };
  __bf16* Wsa_b = (__bf16*)carve((size_t)BEL * KSA * 2);
  __bf16* Wih_b = (__bf16*)carve((size_t)3 * BEL * BEL * 2);
  __bf16* Whh_b = (__bf16*)carve((size_t)3 * BEL * BEL * 2);
  __bf16* Wbp_b = (__bf16*)carve((size_t)HH * BEL * 2);
  __bf16* Wsp_b = (__bf16*)carve((size_t)2 * SS * HH * 2);
  __bf16* Wbq_b = (__bf16*)carve((size_t)HH * KBQ * 2);
  __bf16* Wsq_b = (__bf16*)carve((size_t)2 * SS * HH * 2);
  __bf16* Xsa   = (__bf16*)carve((size_t)BB * KSA * 2);
  __bf16* hid_b = (__bf16*)carve((size_t)BB * BEL * 2);
  __bf16* bel_b = (__bf16*)carve((size_t)BB * BEL * 2);
  __bf16* Xbq   = (__bf16*)carve((size_t)BB * KBQ * 2);
  __bf16* hp_b  = (__bf16*)carve((size_t)BB * HH * 2);
  __bf16* hq_b  = (__bf16*)carve((size_t)BB * HH * 2);
  float*  gi_f  = (float*)carve((size_t)BB * 3 * BEL * 4);
  float*  gh_f  = (float*)carve((size_t)BB * 3 * BEL * 4);
  float*  bel_f = (float*)carve((size_t)BB * BEL * 4);
  float*  st_f  = (float*)carve((size_t)BB * SS * 4);
  float*  p_f   = (float*)carve((size_t)BB * 2 * SS * 4);
  float*  q_f   = (float*)carve((size_t)BB * 2 * SS * 4);
  (void)ws_size; (void)in_sizes; (void)n_in; (void)out_size;

  const int TPB = 256;
  auto g1 = [&](int n) { return dim3((n + TPB - 1) / TPB); };

  // ---- one-time per call: weight conversion + state init ----
  {
    int n;
    n = BEL * KSA;       cvt_bf16_k<<<g1(n), TPB, 0, stream>>>(W_sa, Wsa_b, n);
    n = 3 * BEL * BEL;   cvt_bf16_k<<<g1(n), TPB, 0, stream>>>(W_ih, Wih_b, n);
    n = 3 * BEL * BEL;   cvt_bf16_k<<<g1(n), TPB, 0, stream>>>(W_hh, Whh_b, n);
    n = HH * BEL;        cvt_bf16_k<<<g1(n), TPB, 0, stream>>>(W_bp, Wbp_b, n);
    n = 2 * SS * HH;     cvt_bf16_k<<<g1(n), TPB, 0, stream>>>(W_sp, Wsp_b, n);
    n = HH * KBQ;        cvt_bf16_k<<<g1(n), TPB, 0, stream>>>(W_bq, Wbq_b, n);
    n = 2 * SS * HH;     cvt_bf16_k<<<g1(n), TPB, 0, stream>>>(W_sq, Wsq_b, n);
    init_state_k<<<g1(BB * SS), TPB, 0, stream>>>(prev_state, st_f, BB * SS);
    init_belief_k<<<g1(BB * BEL), TPB, 0, stream>>>(prev_bel, bel_f, bel_b, BB * BEL);
  }

  const dim3 blk(128);
  for (int t = 0; t < TT; ++t) {
    const float* act_t = actions + (size_t)t * BB * AA;
    const float* obs_t = obs     + (size_t)t * BB * EE;
    const float* nt_t  = nonterm + (size_t)t * BB;
    const float* me_t  = map_e   + (size_t)t * BB * MM;
    const float* mp_t  = map_ep  + (size_t)t * BB * MM;
    const float* np_t  = noi_p   + (size_t)t * BB * SS;
    const float* nq_t  = noi_q   + (size_t)t * BB * SS;

    // 1) pack [s*nt, a, me] and [*, obs, mp]
    pack_sa_k<<<g1(BB * KSA), TPB, 0, stream>>>(st_f, nt_t, act_t, me_t, Xsa);
    pack_bq_k<<<g1(BB * (EE + MM)), TPB, 0, stream>>>(obs_t, mp_t, Xbq);

    // 2) hid = relu(Xsa @ Wsa^T + b_sa)  [512x1024], bf16 out
    gemm_wmma_k<1, 0, 1><<<dim3(BEL / 128, BB / 32), blk, 0, stream>>>(
        Xsa, Wsa_b, b_sa, nullptr, hid_b, KSA, BEL);

    // 3) gi = hid @ Wih^T + b_ih ; gh = belief @ Whh^T + b_hh  [512x3072] fp32
    gemm_wmma_k<0, 1, 0><<<dim3(3 * BEL / 128, BB / 32), blk, 0, stream>>>(
        hid_b, Wih_b, b_ih, gi_f, nullptr, BEL, 3 * BEL);
    gemm_wmma_k<0, 1, 0><<<dim3(3 * BEL / 128, BB / 32), blk, 0, stream>>>(
        bel_b, Whh_b, b_hh, gh_f, nullptr, BEL, 3 * BEL);

    // 4) GRU gates -> new belief (fp32 + bf16 + X_bq + d_out)
    gru_gates_k<<<g1(BB * BEL), TPB, 0, stream>>>(
        gi_f, gh_f, bel_f, bel_b, Xbq, out_belief + (size_t)t * BB * BEL);

    // 5) prior: hp = relu(belief @ Wbp^T + b_bp); p = hp @ Wsp^T + b_sp
    gemm_wmma_k<1, 0, 1><<<dim3(HH / 128, BB / 32), blk, 0, stream>>>(
        bel_b, Wbp_b, b_bp, nullptr, hp_b, BEL, HH);
    gemm_wmma_k<0, 1, 0><<<dim3(2 * SS / 128, BB / 32), blk, 0, stream>>>(
        hp_b, Wsp_b, b_sp, p_f, nullptr, HH, 2 * SS);
    dist_out_k<0><<<g1(BB * SS), TPB, 0, stream>>>(
        p_f, np_t, out_pmean + (size_t)t * BB * SS, out_pstd + (size_t)t * BB * SS,
        out_pstate + (size_t)t * BB * SS, nullptr);

    // 6) posterior: hq = relu(Xbq @ Wbq^T + b_bq); q = hq @ Wsq^T + b_sq
    gemm_wmma_k<1, 0, 1><<<dim3(HH / 128, BB / 32), blk, 0, stream>>>(
        Xbq, Wbq_b, b_bq, nullptr, hq_b, KBQ, HH);
    gemm_wmma_k<0, 1, 0><<<dim3(2 * SS / 128, BB / 32), blk, 0, stream>>>(
        hq_b, Wsq_b, b_sq, q_f, nullptr, HH, 2 * SS);
    dist_out_k<1><<<g1(BB * SS), TPB, 0, stream>>>(
        q_f, nq_t, out_qmean + (size_t)t * BB * SS, out_qstd + (size_t)t * BB * SS,
        out_qstate + (size_t)t * BB * SS, st_f);
  }
}
